// GCNGraphSage_79972291052242
// MI455X (gfx1250) — matrix operand order repaired and checked by reading, not compiled, not run on hardware
//
#include <hip/hip_runtime.h>
#include <hip/hip_bf16.h>

typedef _Float16 v16h __attribute__((ext_vector_type(16)));
typedef _Float16 h8   __attribute__((ext_vector_type(8)));
typedef float    v8f  __attribute__((ext_vector_type(8)));

#define D_FEAT 128
#define UNITS  256

// K1: deg[i] = 1.0f (self-loop weight contribution)
__global__ void k_deg_init(float* __restrict__ deg, int n) {
    int i = blockIdx.x * blockDim.x + threadIdx.x;
    if (i < n) deg[i] = 1.0f;
}

// K2: deg[row[e]] += w[e]  (segment_sum over row indices)
__global__ void k_deg_acc(const int* __restrict__ row, const float* __restrict__ w,
                          float* __restrict__ deg, int n_edges) {
    int e = blockIdx.x * blockDim.x + threadIdx.x;
    if (e < n_edges) atomicAdd(deg + row[e], w[e]);
}

// K3: dinv = rsqrt(deg) (in-place), and self-loop init agg[i][:] = dinv^2 * x[i][:]
// One wave (32 lanes) per node, float4 per lane covers 128 features.
__global__ void k_node_init(const float* __restrict__ x, float* __restrict__ deg_dinv,
                            float* __restrict__ agg, int n) {
    int idx  = blockIdx.x * blockDim.x + threadIdx.x;
    int node = idx >> 5;
    int lane = idx & 31;
    if (node >= n) return;
    float d  = deg_dinv[node];                 // all 32 lanes read before lane 0 writes
    float di = (d > 0.f) ? rsqrtf(d) : 0.f;
    if (lane == 0) deg_dinv[node] = di;
    float s = di * di;
    float4 v = ((const float4*)(x + (size_t)node * D_FEAT))[lane];
    float4 o; o.x = s * v.x; o.y = s * v.y; o.z = s * v.z; o.w = s * v.w;
    ((float4*)(agg + (size_t)node * D_FEAT))[lane] = o;
}

// K4: wave-per-edge weighted scatter-add: agg[row] += dinv[row]*w*dinv[col] * x[col]
// x and agg are L2-resident (51.2 MB each vs 192 MB L2) -> atomics resolve in L2.
__global__ void k_edge_agg(const float* __restrict__ x, const float* __restrict__ w,
                           const int* __restrict__ row, const int* __restrict__ col,
                           const float* __restrict__ dinv, float* __restrict__ agg,
                           int n_edges) {
    int lane = threadIdx.x & 31;
    int e = blockIdx.x * (blockDim.x >> 5) + (threadIdx.x >> 5);
    if (e >= n_edges) return;
    int r = row[e], c = col[e];
    float coef = dinv[r] * w[e] * dinv[c];
    float4 v = ((const float4*)(x + (size_t)c * D_FEAT))[lane];
    float* dst = agg + (size_t)r * D_FEAT + lane * 4;
    atomicAdd(dst + 0, coef * v.x);
    atomicAdd(dst + 1, coef * v.y);
    atomicAdd(dst + 2, coef * v.z);
    atomicAdd(dst + 3, coef * v.w);
}

// K5: convert+transpose weight matrix: Bt[n][k] = (f16) kernel[k][n]  (column-major f16)
__global__ void k_convB(const float* __restrict__ ker, _Float16* __restrict__ Bt, int total) {
    int i = blockIdx.x * blockDim.x + threadIdx.x;
    if (i < total) {
        int k = i >> 8;         // i = k*256 + n
        int n = i & (UNITS - 1);
        Bt[(size_t)n * D_FEAT + k] = (_Float16)ker[i];
    }
}

// K6: WMMA GEMM: out = relu(agg(Mx128) * W(128x256) + bias)
// Block = 512 threads = 16 waves; block owns 16 rows; wave w owns N-tile [16w, 16w+16).
// K=128 -> 4x v_wmma_f32_16x16x32_f16 per wave, f32 accumulate.
__global__ void __launch_bounds__(512)
k_gemm_wmma(const float* __restrict__ agg, const _Float16* __restrict__ Bt,
            const float* __restrict__ bias, float* __restrict__ out, int M) {
    __shared__ _Float16 As[16][D_FEAT];   // 4 KB: A strip converted to f16
    int m0 = blockIdx.x * 16;
    bool full = (m0 + 16 <= M);           // M % 16 == 0 for this problem -> always true

    if (full) {
        for (int i = threadIdx.x; i < 16 * D_FEAT; i += blockDim.x) {
            int r = i >> 7, c = i & (D_FEAT - 1);
            As[r][c] = (_Float16)agg[(size_t)(m0 + r) * D_FEAT + c];
        }
    } else {
        for (int i = threadIdx.x; i < 16 * D_FEAT; i += blockDim.x) {
            int r = i >> 7, c = i & (D_FEAT - 1);
            int gr = m0 + r;
            As[r][c] = (_Float16)((gr < M) ? agg[(size_t)gr * D_FEAT + c] : 0.f);
        }
    }
    __syncthreads();

    int lane = threadIdx.x & 31;
    int wv   = threadIdx.x >> 5;      // 0..15
    int n0   = wv * 16;

    v8f acc = {};
    #pragma unroll
    for (int kk = 0; kk < D_FEAT; kk += 32) {
        // A fragment (16x32 f16): lanes 0-15 row=lane, K=kk..kk+15;
        //                          lanes 16-31 row=lane-16, K=kk+16..kk+31.
        union { v16h v; h8 h[2]; } a, b;
        int ar = lane & 15;
        int ak = kk + ((lane < 16) ? 0 : 16);
        a.h[0] = *(const h8*)&As[ar][ak];
        a.h[1] = *(const h8*)&As[ar][ak + 8];
        // B fragment (32x16 f16): lane holds column n0+(lane&15);
        // lanes 0-15 K=kk..kk+15, lanes 16-31 K=kk+16..kk+31 (contiguous in col-major Bt).
        int bc = n0 + (lane & 15);
        int bk = kk + ((lane < 16) ? 0 : 16);
        const h8* bp = (const h8*)(Bt + (size_t)bc * D_FEAT + bk);
        b.h[0] = bp[0];
        b.h[1] = bp[1];
        acc = __builtin_amdgcn_wmma_f32_16x16x32_f16(
            /*neg_a=*/false, a.v, /*neg_b=*/false, b.v,
            /*c_mod=*/(short)0, acc, /*reuse_a=*/false, /*reuse_b=*/false);
    }

    // C/D layout: VGPR i, lanes 0-15: M=m0+i, N=n0+lane; lanes 16-31: M=m0+8+i, N=n0+lane-16
    int ccol  = n0 + (lane & 15);
    float bv  = bias[ccol];
    int rbase = m0 + ((lane < 16) ? 0 : 8);
    float* outp = out + (size_t)rbase * UNITS + ccol;

    if (full) {
        // Common path: 8 unconditional strided stores (no exec-mask churn).
        #pragma unroll
        for (int i = 0; i < 8; ++i) {
            float v = acc[i] + bv;
            outp[(size_t)i * UNITS] = (v > 0.f) ? v : 0.f;
        }
    } else {
        #pragma unroll
        for (int i = 0; i < 8; ++i) {
            if (rbase + i < M) {
                float v = acc[i] + bv;
                outp[(size_t)i * UNITS] = (v > 0.f) ? v : 0.f;
            }
        }
    }
}

extern "C" void kernel_launch(void* const* d_in, const int* in_sizes, int n_in,
                              void* d_out, int out_size, void* d_ws, size_t ws_size,
                              hipStream_t stream) {
    const float* x    = (const float*)d_in[0];
    const float* ew   = (const float*)d_in[1];
    const float* ker  = (const float*)d_in[2];
    const float* bias = (const float*)d_in[3];
    const int*   ei   = (const int*)d_in[4];

    int n_nodes = in_sizes[0] / D_FEAT;
    int n_edges = in_sizes[1];
    const int* row = ei;
    const int* col = ei + n_edges;
    float* out = (float*)d_out;

    // Workspace layout: dinv (n f32) | Bt (128*256 f16) | agg (n*128 f32)
    char* ws = (char*)d_ws;
    size_t off = 0;
    float* deg_dinv = (float*)(ws + off);
    off += ((size_t)n_nodes * sizeof(float) + 255) & ~(size_t)255;
    _Float16* Bt = (_Float16*)(ws + off);
    off += ((size_t)D_FEAT * UNITS * sizeof(_Float16) + 255) & ~(size_t)255;
    float* agg = (float*)(ws + off);

    k_deg_init <<<(n_nodes + 255) / 256, 256, 0, stream>>>(deg_dinv, n_nodes);
    k_deg_acc  <<<(n_edges + 255) / 256, 256, 0, stream>>>(row, ew, deg_dinv, n_edges);
    k_node_init<<<((n_nodes * 32) + 255) / 256, 256, 0, stream>>>(x, deg_dinv, agg, n_nodes);
    k_edge_agg <<<(n_edges + 7) / 8, 256, 0, stream>>>(x, ew, row, col, deg_dinv, agg, n_edges);
    k_convB    <<<(D_FEAT * UNITS + 255) / 256, 256, 0, stream>>>(ker, Bt, D_FEAT * UNITS);
    k_gemm_wmma<<<(n_nodes + 15) / 16, 512, 0, stream>>>(agg, Bt, bias, out, n_nodes);
}
